// Model_57784490000658
// MI455X (gfx1250) — matrix operand order repaired and checked by reading, not compiled, not run on hardware
//
#include <hip/hip_runtime.h>
#include <hip/hip_bf16.h>
#include <math.h>

// ---------------------------------------------------------------------------
// Fused hyperspectral VQ unmixing forward for gfx1250 (MI455X, wave32, WMMA).
// Strategy: per-block tile of 64 pixels; entire enc_z / enc_a / VQ / decoder
// chain computed from LDS with v_wmma_f32_16x16x32_f16; BN folded into conv
// weights (f16) by a prep pass; y_hat stores staged through LDS for coalescing.
// ---------------------------------------------------------------------------

typedef __attribute__((ext_vector_type(16))) _Float16 v16h;
typedef __attribute__((ext_vector_type(8)))  float    v8f;

#define P_DIM 4
#define L_DIM 224
#define B_DIM 4
#define H_DIM 256
#define W_DIM 256
#define HW    (H_DIM * W_DIM)
#define NPIX  (B_DIM * HW)
#define MTILE 64
#define NTHREADS 128   // 4 wave32s

union FragU { unsigned u[8]; v16h h; };

// A-matrix (16x32 f16) K-pair base for VGPR v, lane-half h (ISA 7.12.2 layout)
__device__ __forceinline__ int a_kpair(int v, int half) {
    return ((v < 4) ? (2 * v) : (16 + 2 * (v - 4))) + half * 8;
}

// One M=16 strip (this wave) x N outputs, K from LDS row-major [pix][K] f16,
// weights f16 [N][K] row-major in global (L2-resident), f32 bias, optional ReLU.
template <typename OutT>
__device__ __forceinline__ void gemm_tile(
    const _Float16* __restrict__ sIn, int K,
    const _Float16* __restrict__ gW, const float* __restrict__ gB,
    int N, OutT* __restrict__ sOut, int ldOut, bool relu,
    int lane, int wavebase)
{
    const int half = lane >> 4;
    const int l15  = lane & 15;
    const int arow = wavebase + l15;
    const int KC = (K + 31) >> 5;
    const int NT = (N + 15) >> 4;
    for (int nt = 0; nt < NT; ++nt) {
        const int nb = nt << 4;
        v8f acc = {0.f, 0.f, 0.f, 0.f, 0.f, 0.f, 0.f, 0.f};
        for (int kc = 0; kc < KC; ++kc) {
            const int kb = kc << 5;
            FragU A, Bf;
#pragma unroll
            for (int v = 0; v < 8; ++v) {
                const int ka = kb + a_kpair(v, half);
                unsigned av = 0u;
                if (ka < K) av = *(const unsigned*)(sIn + arow * K + ka);
                A.u[v] = av;
                const int kbb = kb + a_kpair(v, half);
                const int col = nb + l15;
                unsigned bv = 0u;
                if (col < N && kbb < K)
                    bv = *(const unsigned*)(gW + (size_t)col * K + kbb);
                Bf.u[v] = bv;
            }
            acc = __builtin_amdgcn_wmma_f32_16x16x32_f16(
                false, A.h, false, Bf.h, (short)0, acc, false, false);
        }
#pragma unroll
        for (int r = 0; r < 8; ++r) {
            const int m = wavebase + r + half * 8;
            const int n = nb + l15;
            if (n < N) {
                float val = acc[r] + gB[n];
                if (relu && val < 0.f) val = 0.f;
                sOut[m * ldOut + n] = (OutT)val;
            }
        }
    }
}

struct KArgs {
    const float* x;                               // [B,L,H,W] f32
    const float* emb;                             // [224][4]  f32
    const float* wd0; const float* bd0;           // dec conv1 f32 (K=4, VALU)
    const _Float16 *wz0, *wz1, *wz2, *wz3;        // folded enc_z weights f16
    const _Float16 *wa0, *wa1, *wa2, *wa3, *wa4;  // folded enc_a weights f16
    const _Float16 *wd1;                          // dec conv2 weights f16
    const float *bz0, *bz1, *bz2, *bz3;
    const float *ba0, *ba1, *ba2, *ba3, *ba4;
    const float *bd1;
    float* outY; float* outA; float* outVQ;
};

__global__ __launch_bounds__(NTHREADS)
void fused_vq_unmix(KArgs a) {
    __shared__ _Float16 sX[MTILE * L_DIM];   // 28672 B: input tile, reused as sigmoid(h)
    __shared__ _Float16 sA[MTILE * 128];     // 16384 B
    __shared__ float    sYB[MTILE * 32];     //  8192 B: y chunk; aliased as sB f16 [64][64]
    __shared__ float    sZ[MTILE * P_DIM];   //  1024 B
    __shared__ float    sL[MTILE * P_DIM];   //  1024 B
    __shared__ float    sQ[MTILE * P_DIM];   //  1024 B
    __shared__ float    sRed[NTHREADS];      //   512 B   (total 56832 B)
    _Float16* sB = (_Float16*)sYB;

    const int tid      = threadIdx.x;
    const int lane     = tid & 31;
    const int wavebase = (tid >> 5) << 4;
    const int pix0     = blockIdx.x * MTILE;
    const int b        = pix0 / HW;
    const int rem      = pix0 - b * HW;      // h*W + w (tile stays inside one row)

    // ---- load input tile [64 pix][224 ch] as f16 (coalesced 64-float runs) ----
    for (int i = tid; i < MTILE * L_DIM; i += NTHREADS) {
        const int c = i >> 6, p = i & 63;
        sX[p * L_DIM + c] =
            (_Float16)a.x[(size_t)(b * L_DIM + c) * HW + rem + p];
    }
    __syncthreads();

    // ---- encoder_z: 224->128->64->16->4 (BN+ReLU folded) ----
    gemm_tile<_Float16>(sX, 224, a.wz0, a.bz0, 128, sA, 128, true,  lane, wavebase);
    __syncthreads();
    gemm_tile<_Float16>(sA, 128, a.wz1, a.bz1,  64, sB,  64, true,  lane, wavebase);
    __syncthreads();
    gemm_tile<_Float16>(sB,  64, a.wz2, a.bz2,  16, sA,  16, true,  lane, wavebase);
    __syncthreads();
    gemm_tile<float>  (sA,  16, a.wz3, a.bz3,   4, sZ,   4, false, lane, wavebase);
    __syncthreads();

    // ---- encoder_a: 224->128->64->16->16->4 ----
    gemm_tile<_Float16>(sX, 224, a.wa0, a.ba0, 128, sA, 128, true,  lane, wavebase);
    __syncthreads();
    gemm_tile<_Float16>(sA, 128, a.wa1, a.ba1,  64, sB,  64, true,  lane, wavebase);
    __syncthreads();
    gemm_tile<_Float16>(sB,  64, a.wa2, a.ba2,  16, sA,  16, true,  lane, wavebase);
    __syncthreads();
    gemm_tile<_Float16>(sA,  16, a.wa3, a.ba3,  16, sB,  16, true,  lane, wavebase);
    __syncthreads();
    gemm_tile<float>  (sB,  16, a.wa4, a.ba4,   4, sL,   4, false, lane, wavebase);
    __syncthreads();

    // ---- channel softmax -> a_hat (coalesced per-channel stores) ----
    if (tid < MTILE) {
        const float v0 = sL[tid * 4 + 0], v1 = sL[tid * 4 + 1];
        const float v2 = sL[tid * 4 + 2], v3 = sL[tid * 4 + 3];
        float mx = fmaxf(fmaxf(v0, v1), fmaxf(v2, v3));
        const float e0 = __expf(v0 - mx), e1 = __expf(v1 - mx);
        const float e2 = __expf(v2 - mx), e3 = __expf(v3 - mx);
        const float inv = 1.f / (e0 + e1 + e2 + e3);
        a.outA[(size_t)(b * 4 + 0) * HW + rem + tid] = e0 * inv;
        a.outA[(size_t)(b * 4 + 1) * HW + rem + tid] = e1 * inv;
        a.outA[(size_t)(b * 4 + 2) * HW + rem + tid] = e2 * inv;
        a.outA[(size_t)(b * 4 + 3) * HW + rem + tid] = e3 * inv;
    }

    // ---- VQ: raw reshape(-1,4) of [B,4,H,W] => vectors are 4 consecutive W
    //      positions of one channel. 64 vectors per tile (4 ch * 16 groups). ----
    if (tid < MTILE) {
        const int c = tid >> 4;       // channel
        const int g = tid & 15;       // group of 4 pixels
        const int p0 = g * 4;
        const float z0 = sZ[(p0 + 0) * 4 + c], z1 = sZ[(p0 + 1) * 4 + c];
        const float z2 = sZ[(p0 + 2) * 4 + c], z3 = sZ[(p0 + 3) * 4 + c];
        int best = 0; float bd = 3.4e38f;
        for (int l = 0; l < L_DIM; ++l) {
            const float d0 = z0 - a.emb[l * 4 + 0];
            const float d1 = z1 - a.emb[l * 4 + 1];
            const float d2 = z2 - a.emb[l * 4 + 2];
            const float d3 = z3 - a.emb[l * 4 + 3];
            const float d = d0 * d0 + d1 * d1 + d2 * d2 + d3 * d3;
            if (d < bd) { bd = d; best = l; }
        }
        const float q0 = a.emb[best * 4 + 0], q1 = a.emb[best * 4 + 1];
        const float q2 = a.emb[best * 4 + 2], q3 = a.emb[best * 4 + 3];
        sQ[(p0 + 0) * 4 + c] = q0; sQ[(p0 + 1) * 4 + c] = q1;
        sQ[(p0 + 2) * 4 + c] = q2; sQ[(p0 + 3) * 4 + c] = q3;
        const float l0 = q0 - z0, l1 = q1 - z1, l2 = q2 - z2, l3 = q3 - z3;
        sRed[tid] = l0 * l0 + l1 * l1 + l2 * l2 + l3 * l3;
    } else {
        sRed[tid] = 0.f;
    }
    __syncthreads();
    for (int s = NTHREADS / 2; s > 0; s >>= 1) {
        if (tid < s) sRed[tid] += sRed[tid + s];
        __syncthreads();
    }
    if (tid == 0)
        atomicAdd(a.outVQ, sRed[0] * ((1.0f + 0.9f) / (float)(B_DIM * P_DIM * HW)));
    __syncthreads();

    // ---- decoder conv1 (K=4, VALU) + sigmoid -> sX reused as h [64][224] f16 ----
    for (int i = tid; i < MTILE * L_DIM; i += NTHREADS) {
        const int o = i >> 6, p = i & 63;
        float s = a.bd0[o];
#pragma unroll
        for (int j = 0; j < 4; ++j) s += sQ[p * 4 + j] * a.wd0[o * 4 + j];
        sX[p * L_DIM + o] = (_Float16)(1.f / (1.f + __expf(-s)));
    }
    __syncthreads();

    // ---- decoder conv2: 224->224 WMMA, staged through LDS for coalesced y_hat ----
    {
        const int half = lane >> 4;
        const int l15  = lane & 15;
        const int arow = wavebase + l15;
        for (int cn = 0; cn < 7; ++cn) {
#pragma unroll
            for (int t2 = 0; t2 < 2; ++t2) {
                const int nb = (cn * 2 + t2) * 16;
                v8f acc = {0.f, 0.f, 0.f, 0.f, 0.f, 0.f, 0.f, 0.f};
                for (int kc = 0; kc < 7; ++kc) {
                    const int kb = kc * 32;
                    FragU A, Bf;
#pragma unroll
                    for (int v = 0; v < 8; ++v) {
                        A.u[v]  = *(const unsigned*)(sX + arow * L_DIM + kb +
                                                     a_kpair(v, half));
                        Bf.u[v] = *(const unsigned*)(a.wd1 +
                                                     (size_t)(nb + l15) * L_DIM +
                                                     kb + a_kpair(v, half));
                    }
                    acc = __builtin_amdgcn_wmma_f32_16x16x32_f16(
                        false, A.h, false, Bf.h, (short)0, acc, false, false);
                }
#pragma unroll
                for (int r = 0; r < 8; ++r) {
                    const int m = wavebase + r + half * 8;
                    sYB[m * 32 + t2 * 16 + l15] = acc[r] + a.bd1[nb + l15];
                }
            }
            __syncthreads();
            for (int i = tid; i < MTILE * 32; i += NTHREADS) {
                const int n = i >> 6, p = i & 63;
                a.outY[(size_t)(b * L_DIM + cn * 32 + n) * HW + rem + p] =
                    sYB[p * 32 + n];
            }
            __syncthreads();
        }
    }
}

// ---------------------------------------------------------------------------
// Prep: fold eval-mode BN into conv (scale/shift per out-channel), emit f16
// weights [O][C] row-major + f32 bias into workspace. Non-BN: plain convert.
// ---------------------------------------------------------------------------
__global__ void fold_layer(const float* __restrict__ w, const float* __restrict__ b,
                           const float* __restrict__ g, const float* __restrict__ beta,
                           const float* __restrict__ m, const float* __restrict__ v,
                           _Float16* __restrict__ wOut, float* __restrict__ bOut,
                           int C, int hasBN) {
    const int o = blockIdx.x;
    float scale = 1.f, shift = 0.f;
    if (hasBN) {
        const float inv = rsqrtf(v[o] + 1e-5f);
        scale = g[o] * inv;
        shift = beta[o] - m[o] * scale;
    }
    if (threadIdx.x == 0) bOut[o] = b[o] * scale + shift;
    for (int c = threadIdx.x; c < C; c += blockDim.x)
        wOut[(size_t)o * C + c] = (_Float16)(w[(size_t)o * C + c] * scale);
}

// e_hat = dec0 weight (detached) copied to output; also zero vq_loss accumulator.
__global__ void finalize_init(const float* __restrict__ wd0,
                              float* __restrict__ outE, float* __restrict__ outVQ) {
    const int i = blockIdx.x * blockDim.x + threadIdx.x;
    if (i < L_DIM * P_DIM) outE[i] = wd0[i];
    if (i == 0) *outVQ = 0.f;
}

static inline size_t alignUp(size_t x, size_t a) { return (x + a - 1) & ~(a - 1); }

extern "C" void kernel_launch(void* const* d_in, const int* in_sizes, int n_in,
                              void* d_out, int out_size, void* d_ws, size_t ws_size,
                              hipStream_t stream) {
    (void)in_sizes; (void)n_in; (void)out_size; (void)ws_size;
    // JAX pytree flatten order (dict keys sorted):
    // 0:inputs | dec0: 1:b 2:w | dec1: 3:b 4:w | 5:emb
    // enc_a L0..L3 (b,beta,g,m,v,w): 6..11, 12..17, 18..23, 24..29 | L4: 30:b 31:w
    // enc_z L0..L2: 32..37, 38..43, 44..49 | L3: 50:b 51:w
    const float* x    = (const float*)d_in[0];
    const float* bd0  = (const float*)d_in[1];
    const float* wd0  = (const float*)d_in[2];
    const float* bd1f = (const float*)d_in[3];
    const float* wd1f = (const float*)d_in[4];
    const float* emb  = (const float*)d_in[5];

    // workspace bump allocator (f16 weights + f32 biases)
    char* ws = (char*)d_ws;
    size_t off = 0;
    auto takeH = [&](size_t elems) { off = alignUp(off, 256);
        _Float16* p = (_Float16*)(ws + off); off += elems * sizeof(_Float16); return p; };
    auto takeF = [&](size_t elems) { off = alignUp(off, 256);
        float* p = (float*)(ws + off); off += elems * sizeof(float); return p; };

    _Float16 *wz0 = takeH(128 * 224), *wz1 = takeH(64 * 128),
             *wz2 = takeH(16 * 64),   *wz3 = takeH(4 * 16);
    _Float16 *wa0 = takeH(128 * 224), *wa1 = takeH(64 * 128),
             *wa2 = takeH(16 * 64),   *wa3 = takeH(16 * 16), *wa4 = takeH(4 * 16);
    _Float16 *wd1 = takeH(224 * 224);
    float *bz0 = takeF(128), *bz1 = takeF(64), *bz2 = takeF(16), *bz3 = takeF(4);
    float *ba0 = takeF(128), *ba1 = takeF(64), *ba2 = takeF(16), *ba3 = takeF(16),
          *ba4 = takeF(4);
    float *bd1 = takeF(224);

    #define FOLD_BN(O, C, iw, ib, ibeta, ig, im, iv, WO, BO)                         \
        fold_layer<<<O, 64, 0, stream>>>((const float*)d_in[iw], (const float*)d_in[ib], \
            (const float*)d_in[ig], (const float*)d_in[ibeta],                        \
            (const float*)d_in[im], (const float*)d_in[iv], WO, BO, C, 1)
    #define FOLD_NO(O, C, iw, ib, WO, BO)                                            \
        fold_layer<<<O, 64, 0, stream>>>((const float*)d_in[iw], (const float*)d_in[ib], \
            nullptr, nullptr, nullptr, nullptr, WO, BO, C, 0)

    FOLD_BN(128, 224, 37, 32, 33, 34, 35, 36, wz0, bz0);   // enc_z L0
    FOLD_BN( 64, 128, 43, 38, 39, 40, 41, 42, wz1, bz1);   // enc_z L1
    FOLD_BN( 16,  64, 49, 44, 45, 46, 47, 48, wz2, bz2);   // enc_z L2
    FOLD_NO(  4,  16, 51, 50, wz3, bz3);                   // enc_z L3
    FOLD_BN(128, 224, 11,  6,  7,  8,  9, 10, wa0, ba0);   // enc_a L0
    FOLD_BN( 64, 128, 17, 12, 13, 14, 15, 16, wa1, ba1);   // enc_a L1
    FOLD_BN( 16,  64, 23, 18, 19, 20, 21, 22, wa2, ba2);   // enc_a L2
    FOLD_BN( 16,  16, 29, 24, 25, 26, 27, 28, wa3, ba3);   // enc_a L3
    FOLD_NO(  4,  16, 31, 30, wa4, ba4);                   // enc_a L4
    FOLD_NO(224, 224,  4,  3, wd1, bd1);                   // dec conv2

    float* outY  = (float*)d_out;
    float* outA  = outY + (size_t)B_DIM * L_DIM * HW;      // 58,720,256
    float* outE  = outA + (size_t)B_DIM * P_DIM * HW;      //  1,048,576
    float* outVQ = outE + (size_t)L_DIM * P_DIM;           //        896

    finalize_init<<<4, 256, 0, stream>>>(wd0, outE, outVQ);

    KArgs a;
    a.x = x; a.emb = emb; a.wd0 = wd0; a.bd0 = bd0;
    a.wz0 = wz0; a.wz1 = wz1; a.wz2 = wz2; a.wz3 = wz3;
    a.wa0 = wa0; a.wa1 = wa1; a.wa2 = wa2; a.wa3 = wa3; a.wa4 = wa4;
    a.wd1 = wd1;
    a.bz0 = bz0; a.bz1 = bz1; a.bz2 = bz2; a.bz3 = bz3;
    a.ba0 = ba0; a.ba1 = ba1; a.ba2 = ba2; a.ba3 = ba3; a.ba4 = ba4;
    a.bd1 = bd1f; (void)bd1; a.bd1 = bd1;   // use folded copy (identity for no-BN)
    a.outY = outY; a.outA = outA; a.outVQ = outVQ;

    fused_vq_unmix<<<NPIX / MTILE, NTHREADS, 0, stream>>>(a);
}
